// EdgePredictor_73547019976929
// MI455X (gfx1250) — compile-verified, matrix operand order checked by reading, not run on hardware
//
#include <hip/hip_runtime.h>
#include <math.h>

typedef _Float16 half_t;
typedef __attribute__((ext_vector_type(16))) _Float16 v16h;
typedef __attribute__((ext_vector_type(8)))  float    v8f;

// ---------------- problem constants ----------------
#define BB     128
#define SS     4096
#define HH     128
#define NROWS  (BB * SS)                 // 524288 flattened rows
#define ROWS_PER_WG 256                  // 8 waves x 2 tiles x 16 rows
#define NBLOCKS (NROWS / ROWS_PER_WG)    // 2048

// ---------------- LDS layout (bytes): weights only -> exactly 160KB ----------------
#define LDS_W1_OFF   0u        // half [256][256]  W1^T (outch-major, k contiguous) 131072 B
#define LDS_W2_OFF   131072u   // half [64][256]   W2^T                              32768 B
#define LDS_TOTAL    163840u   // 160KB -> two WGs fill the 320KB CDNA5 WGP

// ---------------- weight prep: f32 -> f16, transposed ----------------
// ws halves: [0..65535] = W1T[n][k], [65536..81919] = W2T[m][k]
__global__ void prep_weights_kernel(const float* __restrict__ W1,
                                    const float* __restrict__ W2,
                                    half_t* __restrict__ wsh)
{
    int o = blockIdx.x * blockDim.x + threadIdx.x;
    if (o < 65536) {
        int n = o >> 8, k = o & 255;
        wsh[o] = (half_t)W1[k * 256 + n];
    } else if (o < 81920) {
        int o2 = o - 65536;
        int m = o2 >> 8, k = o2 & 255;
        wsh[o] = (half_t)W2[k * 64 + m];
    }
}

// 16-bit 16x32 fragment layout (ISA 7.12.2): lane half h = lane>>4,
//   element e<8  -> K = 8h + e ; element e>=8 -> K = 16 + 8h + (e-8)
__device__ __forceinline__ v16h load_frag_lds(const half_t* __restrict__ p)
{
    v16h f;
#pragma unroll
    for (int j = 0; j < 8; ++j) { f[j] = p[j]; f[8 + j] = p[16 + j]; }
    return f;
}

__device__ __forceinline__ v16h cvt_frag_global(const float* __restrict__ p)
{
    v16h f;
#pragma unroll
    for (int j = 0; j < 8; ++j) { f[j] = (half_t)p[j]; f[8 + j] = (half_t)p[16 + j]; }
    return f;
}

// D-tiles (out-channels in VGPRs, rows in lanes) of channel groups 2k / 2k+1
// pack (with ReLU) directly into the next layer's B-fragment for K-group k.
__device__ __forceinline__ v16h pack_relu(v8f lo, v8f hi)
{
    v16h f;
#pragma unroll
    for (int j = 0; j < 8; ++j) {
        f[j]     = (half_t)fmaxf(lo[j], 0.0f);
        f[8 + j] = (half_t)fmaxf(hi[j], 0.0f);
    }
    return f;
}

__device__ __forceinline__ v8f bias_init(const float* __restrict__ bp)
{
    v8f c;
#pragma unroll
    for (int j = 0; j < 8; ++j) c[j] = bp[j];
    return c;
}

#define WMMA(A, Bf, C) __builtin_amdgcn_wmma_f32_16x16x32_f16(false, (A), false, (Bf), (short)0, (C), false, false)

__global__ void __launch_bounds__(256, 1)
edge_mlp_kernel(const float*  __restrict__ nodes,   // [B,S,128]
                const float*  __restrict__ newn,    // [B,1,128]
                const half_t* __restrict__ wsh,     // prepped W1T/W2T f16
                const float*  __restrict__ b1,      // [256]
                const float*  __restrict__ b2,      // [64]
                const float*  __restrict__ w3,      // [64]
                const float*  __restrict__ b3,      // [1]
                float*        __restrict__ out)     // [B*S]
{
    extern __shared__ __align__(16) char smem[];
    half_t* lw1 = (half_t*)(smem + LDS_W1_OFF);
    half_t* lw2 = (half_t*)(smem + LDS_W2_OFF);

    const int wave = threadIdx.x >> 5;
    const int lane = threadIdx.x & 31;
    const int r    = lane & 15;     // row index within tile (N dimension / B-frag column)
    const int h    = lane >> 4;     // lane half -> K sub-range

    const int rowBase = blockIdx.x * ROWS_PER_WG + wave * 32;   // 2 tiles of 16 rows
    const int b       = (blockIdx.x * ROWS_PER_WG) >> 12;       // whole WG in same batch b
    const float* nrow = newn + (size_t)b * HH;

    // ---- kick off the weight fill with CDNA5 async global->LDS copies (ASYNCcnt) ----
    // 163840 B = 10240 x b128; 40 per thread. No VGPR round-trip.
    {
#pragma unroll 4
        for (int i = threadIdx.x; i < 10240; i += 256) {
            int off = i * 16;                         // same byte offset in LDS and in ws
            asm volatile("global_load_async_to_lds_b128 %0, %1, %2"
                         :: "v"(off), "v"(off), "s"(wsh) : "memory");
        }
    }

    // ---- overlap: build X B-fragments from global while the async fill runs ----
    // first 128 feats per-row; last 128 (new_nodes) broadcast -> shared fragments
    v16h x[2][4];
#pragma unroll
    for (int t = 0; t < 2; ++t) {
        const float* xr = nodes + (size_t)(rowBase + t * 16 + r) * HH;
#pragma unroll
        for (int ks = 0; ks < 4; ++ks) x[t][ks] = cvt_frag_global(xr + ks * 32 + 8 * h);
    }
    v16h xn[4];
#pragma unroll
    for (int ks = 0; ks < 4; ++ks) xn[ks] = cvt_frag_global(nrow + ks * 32 + 8 * h);

    asm volatile("s_wait_asynccnt 0x0" ::: "memory");
    __syncthreads();

    // ================= layer 1: h1 = relu(W1^T . X^T + b1) =================
    // ks2 produces h1 channel group [32*ks2, 32*ks2+32) directly as layer-2 B-frags.
    v16h h1f[2][8];
#pragma unroll 1
    for (int ks2 = 0; ks2 < 8; ++ks2) {
        const half_t* wp0 = lw1 + (size_t)((2 * ks2) * 16 + r) * 256 + 8 * h;
        const half_t* wp1 = wp0 + 16 * 256;
        // bias depends on output channel = VGPR index: b1[nt*16 + 8h + i]
        v8f c00 = bias_init(b1 + ks2 * 32 + 8 * h);
        v8f c01 = bias_init(b1 + ks2 * 32 + 16 + 8 * h);
        v8f c10 = c00, c11 = c01;
        v16h wa0 = load_frag_lds(wp0);
        v16h wa1 = load_frag_lds(wp1);
#pragma unroll
        for (int ks = 0; ks < 8; ++ks) {
            v16h na0, na1;
            if (ks < 7) {                      // prefetch next K-step (compile-time)
                na0 = load_frag_lds(wp0 + (ks + 1) * 32);
                na1 = load_frag_lds(wp1 + (ks + 1) * 32);
            }
            const v16h bx0 = (ks < 4) ? x[0][ks] : xn[ks - 4];
            const v16h bx1 = (ks < 4) ? x[1][ks] : xn[ks - 4];
            c00 = WMMA(wa0, bx0, c00);
            c01 = WMMA(wa1, bx0, c01);
            c10 = WMMA(wa0, bx1, c10);
            c11 = WMMA(wa1, bx1, c11);
            if (ks < 7) { wa0 = na0; wa1 = na1; }
        }
        h1f[0][ks2] = pack_relu(c00, c01);
        h1f[1][ks2] = pack_relu(c10, c11);
    }

    // ================= layer 2: h2 = relu(W2^T . h1^T + b2) =================
    v16h h2f[2][2];
#pragma unroll 1
    for (int mt = 0; mt < 2; ++mt) {
        const half_t* wq0 = lw2 + (size_t)((2 * mt) * 16 + r) * 256 + 8 * h;
        const half_t* wq1 = wq0 + 16 * 256;
        v8f d00 = bias_init(b2 + mt * 32 + 8 * h);
        v8f d01 = bias_init(b2 + mt * 32 + 16 + 8 * h);
        v8f d10 = d00, d11 = d01;
        v16h wb0 = load_frag_lds(wq0);
        v16h wb1 = load_frag_lds(wq1);
#pragma unroll
        for (int ks2 = 0; ks2 < 8; ++ks2) {
            v16h nb0, nb1;
            if (ks2 < 7) {
                nb0 = load_frag_lds(wq0 + (ks2 + 1) * 32);
                nb1 = load_frag_lds(wq1 + (ks2 + 1) * 32);
            }
            d00 = WMMA(wb0, h1f[0][ks2], d00);
            d01 = WMMA(wb1, h1f[0][ks2], d01);
            d10 = WMMA(wb0, h1f[1][ks2], d10);
            d11 = WMMA(wb1, h1f[1][ks2], d11);
            if (ks2 < 7) { wb0 = nb0; wb1 = nb1; }
        }
        h2f[0][mt] = pack_relu(d00, d01);
        h2f[1][mt] = pack_relu(d10, d11);
    }

    // ================= layer 3: out = sigmoid(W3^T . h2^T + b3) =================
    // W3 (64x1) zero-padded into row 0 of the A operand.
    v16h a3[2];
#pragma unroll
    for (int mt = 0; mt < 2; ++mt) {
        v16h f = {};
        if (r == 0) {
#pragma unroll
            for (int j = 0; j < 8; ++j) {
                f[j]     = (half_t)w3[32 * mt + 8 * h + j];
                f[8 + j] = (half_t)w3[32 * mt + 16 + 8 * h + j];
            }
        }
        a3[mt] = f;
    }
    const float b3v = b3[0];
#pragma unroll
    for (int t = 0; t < 2; ++t) {
        v8f c3 = {0.f, 0.f, 0.f, 0.f, 0.f, 0.f, 0.f, 0.f};
        c3 = WMMA(a3[0], h2f[t][0], c3);
        c3 = WMMA(a3[1], h2f[t][1], c3);
        if (h == 0) {   // D row 0 (the real W3 row) lives in VGPR0 of lanes 0..15
            float y = c3[0] + b3v;
            out[rowBase + t * 16 + r] = 1.0f / (1.0f + __expf(-y));
        }
    }
}

extern "C" void kernel_launch(void* const* d_in, const int* in_sizes, int n_in,
                              void* d_out, int out_size, void* d_ws, size_t ws_size,
                              hipStream_t stream)
{
    (void)in_sizes; (void)n_in; (void)out_size; (void)ws_size;
    const float* nodes = (const float*)d_in[0];
    const float* newn  = (const float*)d_in[1];
    const float* W1    = (const float*)d_in[2];
    const float* b1    = (const float*)d_in[3];
    const float* W2    = (const float*)d_in[4];
    const float* b2    = (const float*)d_in[5];
    const float* W3    = (const float*)d_in[6];
    const float* b3    = (const float*)d_in[7];
    float*  out = (float*)d_out;
    half_t* wsh = (half_t*)d_ws;

    hipFuncSetAttribute(reinterpret_cast<const void*>(edge_mlp_kernel),
                        hipFuncAttributeMaxDynamicSharedMemorySize, (int)LDS_TOTAL);

    prep_weights_kernel<<<320, 256, 0, stream>>>(W1, W2, wsh);
    edge_mlp_kernel<<<NBLOCKS, 256, LDS_TOTAL, stream>>>(nodes, newn, wsh,
                                                         b1, b2, W3, b3, out);
}
